// PointNetFeaturePropagation_10213432230206
// MI455X (gfx1250) — matrix operand order repaired and checked by reading, not compiled, not run on hardware
//
#include <hip/hip_runtime.h>
#include <hip/hip_bf16.h>

typedef __attribute__((ext_vector_type(16))) _Float16 v16h;
typedef __attribute__((ext_vector_type(8)))  float    v8f;
typedef __attribute__((ext_vector_type(4)))  float    v4f;

#define BN_EPS   1e-5f
#define DIST_EPS 1e-8f

// LDS byte offset of a shared-memory pointer: on amdgcn the generic pointer to
// LDS carries the aperture in the high 32 bits and the wave-relative LDS byte
// offset in the low 32 bits (ISA 10.2 aperture mapping).
__device__ __forceinline__ unsigned int lds_off_u32(const void* p) {
    return (unsigned int)(unsigned long long)p;
}

// Async copy of 16 bytes per lane from global to LDS (ASYNCcnt-tracked).
__device__ __forceinline__ void async_g2l_b128(unsigned int lds_byte_off,
                                               const void* gaddr) {
    asm volatile("global_load_async_to_lds_b128 %0, %1, off"
                 :: "v"(lds_byte_off), "v"(gaddr) : "memory");
}
__device__ __forceinline__ void wait_asynccnt0() {
    asm volatile("s_wait_asynccnt 0x0" ::: "memory");
}

// ---------------------------------------------------------------------------
// Stage 0: pre-pack W (Cout=256 x K, f32 row-major) into f16 laid out in WMMA
// B-fragment order: per K-step kk a contiguous 16KB block, element
// [kk][t][lane][e]  <=>  n = t*16 + (lane&15),  k = kk*32 + (lane>=16?16:0)+e.
// This lets GEMM stage B tiles with pure async global->LDS copies.
// ---------------------------------------------------------------------------
__global__ __launch_bounds__(256) void pack_w_kernel(
    const float* __restrict__ W, _Float16* __restrict__ Wp, int K)
{
    const int idx  = blockIdx.x * 256 + threadIdx.x;   // one f16 element
    const int e    = idx & 15;
    const int lane = (idx >> 4) & 31;
    const int t    = (idx >> 9) & 15;
    const int kk   = idx >> 13;
    const int n    = t * 16 + (lane & 15);
    const int k    = kk * 32 + ((lane >> 4) << 4) + e;
    Wp[idx] = (_Float16)W[(size_t)n * K + k];
}

// ---------------------------------------------------------------------------
// Stage 1: 3-NN inverse-distance interpolation.
// One block = 256 query points of one batch; xyz2 (1024x3 f32, 12KB) staged
// into LDS via async global->LDS copies.
// ---------------------------------------------------------------------------
__global__ __launch_bounds__(256) void three_nn_interp_kernel(
    const float* __restrict__ xyz1, const float* __restrict__ xyz2,
    const float* __restrict__ points2, float* __restrict__ interp,
    int N, int S, int C)
{
    __shared__ __align__(16) float sxyz[3072];   // interleaved x,y,z
    const int b   = blockIdx.y;
    const int tid = threadIdx.x;

    {
        const char* gsrc = (const char*)(xyz2 + (size_t)b * S * 3);
        const unsigned int lbase = lds_off_u32(&sxyz[0]);
        const int nchunks = (S * 3) / 4;         // 16B chunks
        for (int ch = tid; ch < nchunks; ch += 256)
            async_g2l_b128(lbase + ch * 16, gsrc + (size_t)ch * 16);
        wait_asynccnt0();
    }
    __syncthreads();

    const int n = blockIdx.x * 256 + tid;
    const float* q = xyz1 + ((size_t)b * N + n) * 3;
    const float px = q[0], py = q[1], pz = q[2];

    float d0 = 3.4e38f, d1 = 3.4e38f, d2 = 3.4e38f;
    int   i0 = 0, i1 = 0, i2 = 0;
    for (int s = 0; s < S; ++s) {
        const float dx = px - sxyz[s * 3 + 0];
        const float dy = py - sxyz[s * 3 + 1];
        const float dz = pz - sxyz[s * 3 + 2];
        const float d = dx * dx + dy * dy + dz * dz;
        if (d < d0)      { d2 = d1; i2 = i1; d1 = d0; i1 = i0; d0 = d; i0 = s; }
        else if (d < d1) { d2 = d1; i2 = i1; d1 = d;  i1 = s; }
        else if (d < d2) { d2 = d;  i2 = s; }
    }
    const float r0 = 1.f / (d0 + DIST_EPS);
    const float r1 = 1.f / (d1 + DIST_EPS);
    const float r2 = 1.f / (d2 + DIST_EPS);
    const float inv = 1.f / (r0 + r1 + r2);
    const float w0 = r0 * inv, w1 = r1 * inv, w2 = r2 * inv;

    const v4f* p0 = (const v4f*)(points2 + ((size_t)b * S + i0) * C);
    const v4f* p1 = (const v4f*)(points2 + ((size_t)b * S + i1) * C);
    const v4f* p2 = (const v4f*)(points2 + ((size_t)b * S + i2) * C);
    v4f* o = (v4f*)(interp + ((size_t)b * N + n) * C);
    for (int c = 0; c < C / 4; ++c)
        o[c] = w0 * p0[c] + w1 * p1[c] + w2 * p2[c];
}

// ---------------------------------------------------------------------------
// Stage 2: WMMA GEMM  Y[M,256] = X[M,K] * W[256,K]^T  (f16 in, f32 acc)
//   X from A0 (ch 0..255) and optionally A1 (ch 256..511), with optional
//   fused BN+ReLU (preScale/preShift) applied per input channel on stage-in.
//   B tiles arrive pre-packed (Wp) and are copied LDS-ward asynchronously.
//   Also accumulates per-output-channel sum / sum-of-squares for BatchNorm.
// Block: 256 threads = 8 waves; tile 128(M) x 256(N); K step 32.
// ---------------------------------------------------------------------------
__global__ __launch_bounds__(256) void gemm_wmma_kernel(
    const float* __restrict__ A0, const float* __restrict__ A1,
    const _Float16* __restrict__ Wp,       // pre-packed, 16KB per K-step
    const float* __restrict__ preScale, const float* __restrict__ preShift,
    float* __restrict__ Y,
    float* __restrict__ sumBuf, float* __restrict__ sqBuf,
    int K)
{
    __shared__ __align__(32) _Float16 ldsA[8 * 32 * 16];   //  8 KB
    __shared__ __align__(32) _Float16 ldsB[16 * 32 * 16];  // 16 KB
    __shared__ float ldsSum[256], ldsSq[256];

    const int tid    = threadIdx.x;
    const int lane   = tid & 31;
    const int w      = tid >> 5;        // wave id 0..7
    const int blockM = blockIdx.x * 128;

    const v8f zero8 = {0.f, 0.f, 0.f, 0.f, 0.f, 0.f, 0.f, 0.f};
    v8f acc[16];
#pragma unroll
    for (int t = 0; t < 16; ++t) acc[t] = zero8;

    const int arow = tid & 127;         // A staging: row within tile
    const int akb  = (tid >> 7) << 4;   // A staging: k sub-offset 0 or 16
    const unsigned int ldsB_base = lds_off_u32(&ldsB[0]);

    const int nk = K >> 5;
    for (int kk = 0; kk < nk; ++kk) {
        const int k0 = kk << 5;
        __syncthreads();   // previous compute done before LDS overwrite

        // ---- stage B tile: async bulk copy of pre-packed fragments ----
        {
            const char* gb = (const char*)(Wp + (size_t)kk * 8192);
            if (kk + 1 < nk) __builtin_prefetch(gb + 16384, 0, 3);
#pragma unroll
            for (int j = 0; j < 4; ++j) {
                const int ch = tid + j * 256;          // 1024 x 16B chunks
                async_g2l_b128(ldsB_base + ch * 16, gb + (size_t)ch * 16);
            }
        }
        // ---- stage A tile (f32 -> f16, optional fused BN+ReLU) ----
        {
            const int kg = k0 + akb;
            const float* src = (kg < 256)
                ? (A0 + (size_t)(blockM + arow) * 256 + kg)
                : (A1 + (size_t)(blockM + arow) * 256 + (kg - 256));
            v4f vv[4];
#pragma unroll
            for (int j = 0; j < 4; ++j) vv[j] = ((const v4f*)src)[j];
#pragma unroll
            for (int i = 0; i < 16; ++i) {
                float x = vv[i >> 2][i & 3];
                if (preScale) {
                    x = fmaxf(x * preScale[kg + i] + preShift[kg + i], 0.f);
                }
                const int kl = akb + i;
                const int ln = (arow & 15) + ((kl >> 4) << 4);
                const int e  = kl & 15;
                ldsA[(((arow >> 4) * 32) + ln) * 16 + e] = (_Float16)x;
            }
        }
        wait_asynccnt0();
        __syncthreads();

        // ---- compute: each wave: 16 rows x 256 cols via 16 WMMA ----
        const v16h a = *(const v16h*)&ldsA[(w * 32 + lane) * 16];
#pragma unroll
        for (int t = 0; t < 16; ++t) {
            const v16h bf = *(const v16h*)&ldsB[(t * 32 + lane) * 16];
            acc[t] = __builtin_amdgcn_wmma_f32_16x16x32_f16(
                false, a, false, bf, (short)0, acc[t], false, false);
        }
    }

    // ---- epilogue: store Y and reduce per-channel sum / sumsq ----
    ldsSum[tid] = 0.f;
    ldsSq[tid]  = 0.f;
    __syncthreads();

    const int col0  = lane & 15;
    const int rbase = blockM + w * 16 + ((lane >> 4) << 3);
#pragma unroll
    for (int t = 0; t < 16; ++t) {
        const int ch = t * 16 + col0;
        float s = 0.f, qq = 0.f;
#pragma unroll
        for (int r = 0; r < 8; ++r) {
            const float v = acc[t][r];
            Y[(size_t)(rbase + r) * 256 + ch] = v;
            s += v; qq += v * v;
        }
        atomicAdd(&ldsSum[ch], s);
        atomicAdd(&ldsSq[ch], qq);
    }
    __syncthreads();
    atomicAdd(&sumBuf[tid], ldsSum[tid]);
    atomicAdd(&sqBuf[tid],  ldsSq[tid]);
}

// ---------------------------------------------------------------------------
// Small helpers: zero stat accumulators, fold BN stats into scale/shift,
// final elementwise BN+ReLU.
// ---------------------------------------------------------------------------
__global__ void zero_stats_kernel(float* p) { p[threadIdx.x] = 0.f; }

__global__ void finalize_bn_kernel(const float* __restrict__ sum,
                                   const float* __restrict__ sq,
                                   const float* __restrict__ g,
                                   const float* __restrict__ b,
                                   float* __restrict__ scale,
                                   float* __restrict__ shift, float invM)
{
    const int c = threadIdx.x;
    const float mu  = sum[c] * invM;
    const float var = sq[c] * invM - mu * mu;
    const float inv = rsqrtf(var + BN_EPS);
    const float sc  = g[c] * inv;
    scale[c] = sc;
    shift[c] = b[c] - mu * sc;
}

__global__ __launch_bounds__(256) void bn_relu_out_kernel(
    const float* __restrict__ Y, const float* __restrict__ scale,
    const float* __restrict__ shift, float* __restrict__ out, long total4)
{
    const long i = (long)blockIdx.x * 256 + threadIdx.x;
    if (i >= total4) return;
    const int ch4 = (int)(i & 63) * 4;    // 256 channels / 4 per vec
    const v4f y  = ((const v4f*)Y)[i];
    const v4f sc = *(const v4f*)(scale + ch4);
    const v4f sh = *(const v4f*)(shift + ch4);
    v4f o;
#pragma unroll
    for (int j = 0; j < 4; ++j) o[j] = fmaxf(y[j] * sc[j] + sh[j], 0.f);
    ((v4f*)out)[i] = o;
}

// ---------------------------------------------------------------------------
extern "C" void kernel_launch(void* const* d_in, const int* in_sizes, int n_in,
                              void* d_out, int out_size, void* d_ws, size_t ws_size,
                              hipStream_t stream)
{
    (void)in_sizes; (void)n_in; (void)out_size; (void)ws_size;
    const float* xyz1    = (const float*)d_in[0];
    const float* xyz2    = (const float*)d_in[1];
    const float* points1 = (const float*)d_in[2];
    const float* points2 = (const float*)d_in[3];
    const float* w1      = (const float*)d_in[4];
    const float* g1      = (const float*)d_in[5];
    const float* b1      = (const float*)d_in[6];
    const float* w2      = (const float*)d_in[7];
    const float* g2      = (const float*)d_in[8];
    const float* b2      = (const float*)d_in[9];

    const int B = 16, N = 4096, S = 1024, C = 256;
    const int M = B * N;

    float* interp = (float*)d_ws;                       // 64 MB
    float* ybuf   = interp + (size_t)M * C;             // 64 MB
    float* st     = ybuf   + (size_t)M * C;             // 8 KB stats
    float *sum1 = st,        *sq1 = st + 256,  *sc1 = st + 512,  *sh1 = st + 768;
    float *sum2 = st + 1024, *sq2 = st + 1280, *sc2 = st + 1536, *sh2 = st + 1792;
    _Float16* w1p = (_Float16*)(st + 2048);             // 256 KB packed f16
    _Float16* w2p = w1p + 256 * 512;                    // 128 KB packed f16

    // One-time weight packs (tiny) + interpolation
    pack_w_kernel<<<512, 256, 0, stream>>>(w1, w1p, 512);
    pack_w_kernel<<<256, 256, 0, stream>>>(w2, w2p, 256);
    three_nn_interp_kernel<<<dim3(N / 256, B), 256, 0, stream>>>(
        xyz1, xyz2, points2, interp, N, S, C);

    // Layer 1: feats = [points1 | interp] (K=512) -> y1, accumulate stats
    zero_stats_kernel<<<1, 512, 0, stream>>>(sum1);
    gemm_wmma_kernel<<<M / 128, 256, 0, stream>>>(
        points1, interp, w1p, nullptr, nullptr, ybuf, sum1, sq1, 512);
    finalize_bn_kernel<<<1, 256, 0, stream>>>(sum1, sq1, g1, b1, sc1, sh1, 1.f / M);

    // Layer 2: BN1+ReLU fused into A-staging of GEMM2 (K=256) -> y2 (reuse interp)
    zero_stats_kernel<<<1, 512, 0, stream>>>(sum2);
    gemm_wmma_kernel<<<M / 128, 256, 0, stream>>>(
        ybuf, nullptr, w2p, sc1, sh1, interp, sum2, sq2, 256);
    finalize_bn_kernel<<<1, 256, 0, stream>>>(sum2, sq2, g2, b2, sc2, sh2, 1.f / M);

    // Final BN2 + ReLU -> d_out
    const long total4 = (long)M * C / 4;
    bn_relu_out_kernel<<<(unsigned)((total4 + 255) / 256), 256, 0, stream>>>(
        interp, sc2, sh2, (float*)d_out, total4);
}
